// AbmilBagClassifier_80401787781685
// MI455X (gfx1250) — matrix operand order repaired and checked by reading, not compiled, not run on hardware
//
#include <hip/hip_runtime.h>
#include <hip/hip_bf16.h>

// Problem constants (from reference): B=16, N=8192, E=512, P=128, C=1
#define B_ 16
#define N_ 8192
#define E_ 512
#define P_ 128

typedef __attribute__((ext_vector_type(16))) __bf16       bf16x16;
typedef __attribute__((ext_vector_type(8)))  float        f32x8;
typedef __attribute__((ext_vector_type(4)))  float        f32x4;
typedef __attribute__((ext_vector_type(4)))  unsigned int u32x4;

union BF16Frag {
  bf16x16 bf;
  u32x4   q2[2];
};

// ---------------------------------------------------------------------------
// Kernel 0 (one-time prep): convert Wv||Wu (fp32, [E,P] each) into a
// transposed bf16 matrix Wt_g[p][k] (p in 0..255, k in 0..511) in workspace.
// This makes the hot kernel's weight staging a raw byte copy (async DMA-able)
// and removes all W conversions from the GEMM loop. 256 KB -> L2 resident.
// ---------------------------------------------------------------------------
__global__ void __launch_bounds__(256)
abmil_prep_w(const float* __restrict__ Wv, const float* __restrict__ Wu,
             unsigned short* __restrict__ Wt_g)
{
  const int i = blockIdx.x * 256 + threadIdx.x;   // 0 .. 256*512-1
  const int p = i >> 9;                           // column (0..255)
  const int k = i & 511;                          // K index (0..511)
  const float f = (p < P_) ? Wv[(size_t)k * P_ + p]
                           : Wu[(size_t)k * P_ + (p - P_)];
  const __bf16 h = (__bf16)f;                     // RTNE, hw convert
  Wt_g[i] = __builtin_bit_cast(unsigned short, h);
}

// ---------------------------------------------------------------------------
// Kernel 1: gated-attention logits.
//   logit[b,n] = (tanh(x@Wv+bv) * sigmoid(x@Wu+bu)) @ Wa + ba     (C == 1)
// Grid: B*16 blocks x 256 threads (8 waves). Each block loops over 4 chunks of
// 128 rows; each wave does a 16-row x 256-col WMMA GEMM (Wv and Wu fused).
// Weight panels are staged into LDS with GLOBAL_LOAD_ASYNC_TO_LDS_B128
// (ASYNCcnt) from the pre-converted bf16 copy.
// ---------------------------------------------------------------------------
__global__ void __launch_bounds__(256)
abmil_attn_logits(const float* __restrict__ x,
                  const unsigned short* __restrict__ Wt_g,
                  const float* __restrict__ bv, const float* __restrict__ bu,
                  const float* __restrict__ Wa, const float* __restrict__ ba,
                  float* __restrict__ logits)
{
  // bf16 weight panel: [col p (0..255)][k_local (0..63)], 32 KB.
  __shared__ __align__(16) unsigned short Wt[256 * 64];

  const int b     = blockIdx.x >> 4;   // bag
  const int grp   = blockIdx.x & 15;   // block-group within bag
  const int tid   = threadIdx.x;
  const int lane  = tid & 31;
  const int wave  = tid >> 5;
  const int lan16 = lane & 15;
  const int hi    = lane >> 4;         // which 16-lane half
  const float ba0 = ba[0];

  // Per-thread async staging addresses (thread owns column p = tid).
  const unsigned lds_base = (unsigned)(uintptr_t)&Wt[tid * 64];
  const unsigned short* gcol = Wt_g + (size_t)tid * E_;

  for (int c = 0; c < 4; ++c) {
    const int chunk   = grp * 4 + c;               // 0..63
    const int rowbase = chunk * 128 + wave * 16;   // this wave's 16 rows
    // A-fragment source row for this lane (M = lane & 15)
    const float* xrow = x + ((size_t)(b * N_) + rowbase + lan16) * E_;

    f32x8 acc[16];                                 // tiles 0..7 = Wv, 8..15 = Wu
    #pragma unroll
    for (int t = 0; t < 16; ++t) acc[t] = (f32x8){};

    for (int kp = 0; kp < 8; ++kp) {               // K panels of 64 over E=512
      __syncthreads();                             // prior panel fully consumed
      {
        // Async DMA: 128 bytes (64 bf16) of this thread's weight column,
        // global (L2-hit) -> LDS, no VGPR round-trip. Tracked by ASYNCcnt.
        const unsigned short* gsrc = gcol + kp * 64;
        #pragma unroll
        for (int j = 0; j < 8; ++j) {
          const unsigned           la = lds_base + j * 16;
          const unsigned long long ga =
              (unsigned long long)(uintptr_t)gsrc + (unsigned)(j * 16);
          asm volatile("global_load_async_to_lds_b128 %0, %1, off"
                       :: "v"(la), "v"(ga) : "memory");
        }
        asm volatile("s_wait_asynccnt 0" ::: "memory");
      }
      __syncthreads();

      #pragma unroll
      for (int ks = 0; ks < 2; ++ks) {
        const int kg = kp * 64 + ks * 32;          // global K of this step
        // Prefetch next x panel chunk while we compute on this one.
        __builtin_prefetch(xrow + kg + 64, 0, 1);
        // A fragment (16x32 bf16): lane<16 holds K {0..7,16..23},
        // lane>=16 holds K {8..15,24..31}.
        const int ca = hi ? 8 : 0;
        f32x4 a0 = *(const f32x4*)(xrow + kg + ca);
        f32x4 a1 = *(const f32x4*)(xrow + kg + ca + 4);
        f32x4 a2 = *(const f32x4*)(xrow + kg + ca + 16);
        f32x4 a3 = *(const f32x4*)(xrow + kg + ca + 20);
        bf16x16 abf;
        #pragma unroll
        for (int i = 0; i < 4; ++i) {
          abf[i]      = (__bf16)a0[i];
          abf[4 + i]  = (__bf16)a1[i];
          abf[8 + i]  = (__bf16)a2[i];
          abf[12 + i] = (__bf16)a3[i];
        }
        // B fragment (32x16 bf16): lane<16 -> col=lane, K {0..7,16..23};
        // lane>=16 -> col=lane-16, K {8..15,24..31}.
        const int cb = hi ? 8 : 0;
        #pragma unroll
        for (int t = 0; t < 16; ++t) {
          const int p = t * 16 + lan16;
          BF16Frag bfr;
          const unsigned short* wp = &Wt[p * 64 + ks * 32 + cb];
          bfr.q2[0] = *(const u32x4*)(wp);
          bfr.q2[1] = *(const u32x4*)(wp + 16);
          acc[t] = __builtin_amdgcn_wmma_f32_16x16x32_bf16(
                       false, abf, false, bfr.bf, (short)0, acc[t],
                       false, false);
        }
      }
    }

    // Epilogue: gating + projection onto Wa (C == 1).
    // C/D layout: lanes 0-15: col=lane, VGPR r -> row r; lanes 16-31: row 8+r.
    float part[8];
    #pragma unroll
    for (int r = 0; r < 8; ++r) part[r] = 0.f;
    #pragma unroll
    for (int t = 0; t < 8; ++t) {
      const int p  = t * 16 + lan16;
      const float bvp = bv[p], bup = bu[p], wap = Wa[p];
      #pragma unroll
      for (int r = 0; r < 8; ++r) {
        // tanh(z) = (e^{2z}-1)/(e^{2z}+1) via fast exp
        const float e2 = __expf(2.f * (acc[t][r] + bvp));
        const float vv = (e2 - 1.f) / (e2 + 1.f);
        const float uu = 1.f / (1.f + __expf(-(acc[t + 8][r] + bup)));
        part[r] += vv * uu * wap;
      }
    }
    // Reduce over the 16 lanes of each half (cols of the tile).
    #pragma unroll
    for (int r = 0; r < 8; ++r) {
      #pragma unroll
      for (int m = 1; m < 16; m <<= 1)
        part[r] += __shfl_xor(part[r], m, 32);
    }
    if (lan16 == 0) {
      #pragma unroll
      for (int r = 0; r < 8; ++r) {
        const int row = rowbase + hi * 8 + r;
        logits[(size_t)b * N_ + row] = part[r] + ba0;
      }
    }
  }
}

// ---------------------------------------------------------------------------
// Kernel 2: masked softmax over the instance dim for each bag; also zero-inits
// the pooled accumulator. Grid: B blocks x 1024 threads.
// ---------------------------------------------------------------------------
__global__ void __launch_bounds__(1024)
abmil_softmax(const float* __restrict__ logits,
              const int* __restrict__ bag_lens,
              float* __restrict__ A_out, float* __restrict__ pooled)
{
  const int b   = blockIdx.x;
  const int tid = threadIdx.x;
  const int len = bag_lens[b];
  const float* l = logits + (size_t)b * N_;
  __shared__ float red[32];

  if (tid < E_) pooled[b * E_ + tid] = 0.f;

  // Pass 1: masked max
  float m = -3.402823466e38f;
  for (int n = tid; n < N_; n += 1024)
    if (n < len) m = fmaxf(m, l[n]);
  #pragma unroll
  for (int o = 1; o < 32; o <<= 1) m = fmaxf(m, __shfl_xor(m, o, 32));
  if ((tid & 31) == 0) red[tid >> 5] = m;
  __syncthreads();
  if (tid < 32) {
    float v = red[tid];
    #pragma unroll
    for (int o = 1; o < 32; o <<= 1) v = fmaxf(v, __shfl_xor(v, o, 32));
    red[tid] = v;
  }
  __syncthreads();
  const float gmax = red[0];
  __syncthreads();

  // Pass 2: masked sum of exp
  float s = 0.f;
  for (int n = tid; n < N_; n += 1024)
    if (n < len) s += __expf(l[n] - gmax);
  #pragma unroll
  for (int o = 1; o < 32; o <<= 1) s += __shfl_xor(s, o, 32);
  if ((tid & 31) == 0) red[tid >> 5] = s;
  __syncthreads();
  if (tid < 32) {
    float v = red[tid];
    #pragma unroll
    for (int o = 1; o < 32; o <<= 1) v += __shfl_xor(v, o, 32);
    red[tid] = v;
  }
  __syncthreads();
  const float inv = 1.f / red[0];

  for (int n = tid; n < N_; n += 1024) {
    const float w = (n < len) ? __expf(l[n] - gmax) * inv : 0.f;
    A_out[(size_t)b * N_ + n] = w;
  }
}

// ---------------------------------------------------------------------------
// Kernel 3: pooled[b, e] = sum_n A[b,n] * x[b,n,e]. Split over 32 N-chunks per
// bag, accumulated with global_atomic_add_f32. Grid: B*32 blocks x 256 thr.
// ---------------------------------------------------------------------------
__global__ void __launch_bounds__(256)
abmil_pool(const float* __restrict__ x, const float* __restrict__ A_out,
           float* __restrict__ pooled)
{
  const int b     = blockIdx.x >> 5;
  const int chunk = blockIdx.x & 31;
  const int tid   = threadIdx.x;
  const int n0    = chunk * (N_ / 32);
  float acc0 = 0.f, acc1 = 0.f;
  for (int i = 0; i < N_ / 32; ++i) {
    const int n = n0 + i;
    const float w = A_out[(size_t)b * N_ + n];
    const float* xr = x + ((size_t)(b * N_) + n) * E_;
    acc0 += w * xr[tid];
    acc1 += w * xr[tid + 256];
  }
  atomicAdd(&pooled[b * E_ + tid], acc0);
  atomicAdd(&pooled[b * E_ + tid + 256], acc1);
}

// ---------------------------------------------------------------------------
extern "C" void kernel_launch(void* const* d_in, const int* in_sizes, int n_in,
                              void* d_out, int out_size, void* d_ws,
                              size_t ws_size, hipStream_t stream)
{
  const float* x        = (const float*)d_in[0];
  const int*   bag_lens = (const int*)  d_in[1];
  const float* Wv       = (const float*)d_in[2];
  const float* bv       = (const float*)d_in[3];
  const float* Wu       = (const float*)d_in[4];
  const float* bu       = (const float*)d_in[5];
  const float* Wa       = (const float*)d_in[6];
  const float* ba       = (const float*)d_in[7];

  float* A_out  = (float*)d_out;                 // [B, N, 1]
  float* pooled = A_out + (size_t)B_ * N_;       // [B, 1, E]

  unsigned short* Wt_g = (unsigned short*)d_ws;  // [256][512] bf16 (256 KB)
  float* logits = (float*)((char*)d_ws + 256 * 512 * sizeof(unsigned short));

  abmil_prep_w<<<dim3((256 * 512) / 256), dim3(256), 0, stream>>>(Wv, Wu, Wt_g);
  abmil_attn_logits<<<dim3(B_ * 16), dim3(256), 0, stream>>>(
      x, Wt_g, bv, bu, Wa, ba, logits);
  abmil_softmax<<<dim3(B_), dim3(1024), 0, stream>>>(
      logits, bag_lens, A_out, pooled);
  abmil_pool<<<dim3(B_ * 32), dim3(256), 0, stream>>>(x, A_out, pooled);
}
